// GIN_2267742732765
// MI455X (gfx1250) — matrix operand order repaired and checked by reading, not compiled, not run on hardware
//
#include <hip/hip_runtime.h>
#include <hip/hip_bf16.h>

typedef __attribute__((ext_vector_type(16))) _Float16 v16h;
typedef __attribute__((ext_vector_type(8)))  float    v8f;

#define NNODES 50000
#define NEDGES 800000
#define DCH    96
#define HW     192
#define NGRAPH 64
#define MTILES (NNODES / 16)          // 3125 (exact)

#define KT1 3                          // 96/32
#define NT1 12                         // 192/16
#define KT2 6                          // 192/32
#define NT2 6                          // 96/16
#define NHALF1 6                       // N-tiles per wave in GEMM1 (split in 2)
#define FRAG_ELEMS 512                 // 32 lanes * 16 f16 per B fragment
#define PACK1_PER_LAYER (KT1 * NT1 * FRAG_ELEMS)   // 18432
#define PACK2_PER_LAYER (KT2 * NT2 * FRAG_ELEMS)   // 18432

// ---------------------------------------------------------------------------
// Repack W1/W2 (f32 row-major [K][N]) into per-lane B-fragment-major f16.
// Dense 16-bit B (32x16) layout (ISA 7.12.4/7.12.5 B tables, wave32):
//   lane L: column n = L&15 ; K = (L>>4)*8 + (e&7) + (e>>3)*16, e=0..15 (2 per VGPR).
// Packed index: ((kt*NT + nt)*32 + lane)*16 + e  -> one aligned 32B load/frag.
// ---------------------------------------------------------------------------
__global__ void pack_weights(const float* __restrict__ W1,
                             const float* __restrict__ W2,
                             _Float16* __restrict__ p1,
                             _Float16* __restrict__ p2) {
  int idx = blockIdx.x * blockDim.x + threadIdx.x;
  const int perLayer = PACK1_PER_LAYER + PACK2_PER_LAYER;
  if (idx >= 3 * perLayer) return;
  int l = idx / perLayer;
  int r = idx % perLayer;
  if (r < PACK1_PER_LAYER) {
    int frag = r / FRAG_ELEMS, w = r % FRAG_ELEMS;
    int lane = w >> 4, e = w & 15;
    int kt = frag / NT1, nt = frag % NT1;
    int k = kt * 32 + (lane >> 4) * 8 + (e & 7) + (e >> 3) * 16;
    int n = nt * 16 + (lane & 15);
    p1[l * PACK1_PER_LAYER + r] = (_Float16)W1[(l * DCH + k) * HW + n];
  } else {
    int r2 = r - PACK1_PER_LAYER;
    int frag = r2 / FRAG_ELEMS, w = r2 % FRAG_ELEMS;
    int lane = w >> 4, e = w & 15;
    int kt = frag / NT2, nt = frag % NT2;
    int k = kt * 32 + (lane >> 4) * 8 + (e & 7) + (e >> 3) * 16;
    int n = nt * 16 + (lane & 15);
    p2[l * PACK2_PER_LAYER + r2] = (_Float16)W2[(l * HW + k) * DCH + n];
  }
}

// agg = (1 + eps[l]) * h   (self term of GINConv, before edge scatter)
__global__ void init_agg(const float* __restrict__ hin,
                         const float* __restrict__ eps, int layer,
                         float* __restrict__ agg) {
  int idx = blockIdx.x * blockDim.x + threadIdx.x;
  if (idx >= NNODES * DCH) return;
  float s = 1.0f + eps[layer];
  agg[idx] = s * hin[idx];
}

// agg[dst] += h[src] over all edges. One thread per (edge, 4-channel group).
// Everything is L2-resident (19.2MB buffers << 192MB L2) -> fp atomics at L2.
__global__ void scatter_edges(const float* __restrict__ hin,
                              const int* __restrict__ ei,
                              float* __restrict__ agg) {
  int t = blockIdx.x * blockDim.x + threadIdx.x;
  if (t >= NEDGES * 24) return;
  int e = t / 24, q = t % 24;
  int src = ei[e];
  int dst = ei[NEDGES + e];
  const float4 v = *(const float4*)(hin + (size_t)src * DCH + q * 4);
  float* a = agg + (size_t)dst * DCH + q * 4;
  unsafeAtomicAdd(a + 0, v.x);
  unsafeAtomicAdd(a + 1, v.y);
  unsafeAtomicAdd(a + 2, v.z);
  unsafeAtomicAdd(a + 3, v.w);
}

// ---------------------------------------------------------------------------
// GEMM1: z = relu(agg[50000x96] @ W1[96x192] + b1)  -> f16 output
// Two waves per 16-row M-tile, each owning 6 N-tiles (48 acc VGPRs: no spill).
// A fragment (16x32 f16, ISA 7.12.2): lane L -> row m=L&15,
//   K = {kb..kb+7, kb+16..kb+23}, kb = (L>>4)*8.
// ---------------------------------------------------------------------------
__global__ void __launch_bounds__(256)
gin_gemm1(const float* __restrict__ zin,
          const _Float16* __restrict__ wp,
          const float* __restrict__ bias,
          _Float16* __restrict__ zout) {
  int wave = (int)((blockIdx.x * blockDim.x + threadIdx.x) >> 5);
  if (wave >= MTILES * 2) return;             // wave-uniform: EXEC stays all-1s
  int mt = wave >> 1;
  int nbase = (wave & 1) * NHALF1;            // 0 or 6
  int lane = threadIdx.x & 31;
  int mhalf = lane >> 4, mlo = lane & 15;
  const float* arow = zin + (size_t)(mt * 16 + mlo) * DCH;
  const v16h* bfr = ((const v16h*)wp) + lane;

  v8f acc[NHALF1] = {};
#pragma unroll
  for (int kt = 0; kt < KT1; kt++) {
    int kb = kt * 32 + mhalf * 8;
    v16h a;
#pragma unroll
    for (int i = 0; i < 8; i++) {
      a[i]     = (_Float16)arow[kb + i];
      a[i + 8] = (_Float16)arow[kb + 16 + i];
    }
#pragma unroll
    for (int nt = 0; nt < NHALF1; nt++) {
      v16h b = bfr[(kt * NT1 + nbase + nt) * 32];
      acc[nt] = __builtin_amdgcn_wmma_f32_16x16x32_f16(
          false, a, false, b, (short)0, acc[nt], false, false);
    }
  }
  // D layout: lane L, VGPR v -> m = v + 8*(L>>4), n = L&15
#pragma unroll
  for (int nt = 0; nt < NHALF1; nt++) {
    int n = (nbase + nt) * 16 + mlo;
    float bn = bias[n];
#pragma unroll
    for (int v = 0; v < 8; v++) {
      float val = acc[nt][v] + bn;
      val = val > 0.0f ? val : 0.0f;
      int m = mt * 16 + v + mhalf * 8;
      zout[(size_t)m * HW + n] = (_Float16)val;
    }
  }
}

// GEMM2: h = relu(z[50000x192] @ W2[192x96] + b2) -> f32 output
// One wave per M-tile, 6 N-tiles (48 acc VGPRs).
__global__ void __launch_bounds__(256)
gin_gemm2(const _Float16* __restrict__ zin,
          const _Float16* __restrict__ wp,
          const float* __restrict__ bias,
          float* __restrict__ hout) {
  int wave = (int)((blockIdx.x * blockDim.x + threadIdx.x) >> 5);
  if (wave >= MTILES) return;
  int lane = threadIdx.x & 31;
  int mhalf = lane >> 4, mlo = lane & 15;
  const _Float16* arow = zin + (size_t)(wave * 16 + mlo) * HW;
  const v16h* bfr = ((const v16h*)wp) + lane;

  v8f acc[NT2] = {};
#pragma unroll
  for (int kt = 0; kt < KT2; kt++) {
    int kb = kt * 32 + mhalf * 8;
    v16h a;
#pragma unroll
    for (int i = 0; i < 8; i++) {
      a[i]     = arow[kb + i];
      a[i + 8] = arow[kb + 16 + i];
    }
#pragma unroll
    for (int nt = 0; nt < NT2; nt++) {
      v16h b = bfr[(kt * NT2 + nt) * 32];
      acc[nt] = __builtin_amdgcn_wmma_f32_16x16x32_f16(
          false, a, false, b, (short)0, acc[nt], false, false);
    }
  }
#pragma unroll
  for (int nt = 0; nt < NT2; nt++) {
    int n = nt * 16 + mlo;
    float bn = bias[n];
#pragma unroll
    for (int v = 0; v < 8; v++) {
      float val = acc[nt][v] + bn;
      val = val > 0.0f ? val : 0.0f;
      int m = wave * 16 + v + mhalf * 8;
      hout[(size_t)m * DCH + n] = val;
    }
  }
}

__global__ void zero_g(float* __restrict__ g) {
  int i = blockIdx.x * blockDim.x + threadIdx.x;
  if (i < NGRAPH * DCH) g[i] = 0.0f;
}

// global_add_pool: batch is sorted -> per-block LDS accumulation over a
// 128-node chunk, then only (gmax-gmin+1)*96 global atomics per block.
#define POOL_NB 128
__global__ void pool_graphs(const float* __restrict__ h,
                            const int* __restrict__ batch,
                            float* __restrict__ g) {
  __shared__ float acc[NGRAPH * DCH];          // 24 KB of 320 KB LDS
  int tid = threadIdx.x;
  for (int i = tid; i < NGRAPH * DCH; i += 256) acc[i] = 0.0f;
  __syncthreads();

  int base = blockIdx.x * POOL_NB;
  int nn = NNODES - base; if (nn > POOL_NB) nn = POOL_NB;
  int items = nn * DCH;
  for (int i = tid; i < items; i += 256) {
    int node = base + i / DCH;
    int c = i % DCH;
    atomicAdd(&acc[batch[node] * DCH + c], h[(size_t)node * DCH + c]);  // ds_add_f32
  }
  __syncthreads();

  int gmin = batch[base];
  int gmax = batch[base + nn - 1];
  int span = (gmax - gmin + 1) * DCH;
  int off = gmin * DCH;
  for (int i = tid; i < span; i += 256)
    unsafeAtomicAdd(&g[off + i], acc[off + i]);
}

// final_mlp: [64x96] -> relu(g@fW1+fb1) -> @fW2+fb2 -> [64]
__global__ void final_mlp(const float* __restrict__ g,
                          const float* __restrict__ fW1,
                          const float* __restrict__ fb1,
                          const float* __restrict__ fW2,
                          const float* __restrict__ fb2,
                          float* __restrict__ out) {
  __shared__ float hid[NGRAPH * DCH];
  int j = threadIdx.x;
  if (j < DCH) {
    for (int i = 0; i < NGRAPH; i++) {
      float s = fb1[j];
      for (int k = 0; k < DCH; k++) s += g[i * DCH + k] * fW1[k * DCH + j];
      hid[i * DCH + j] = s > 0.0f ? s : 0.0f;
    }
  }
  __syncthreads();
  if (j < NGRAPH) {
    float s = fb2[0];
    for (int k = 0; k < DCH; k++) s += hid[j * DCH + k] * fW2[k];
    out[j] = s;
  }
}

static inline size_t align256(size_t v) { return (v + 255) & ~(size_t)255; }

extern "C" void kernel_launch(void* const* d_in, const int* in_sizes, int n_in,
                              void* d_out, int out_size, void* d_ws, size_t ws_size,
                              hipStream_t stream) {
  const float* x      = (const float*)d_in[0];
  const int*   ei     = (const int*)d_in[1];   // [2, NEDGES]
  const int*   batch  = (const int*)d_in[2];
  const float* W1     = (const float*)d_in[3];
  const float* b1     = (const float*)d_in[4];
  const float* W2     = (const float*)d_in[5];
  const float* b2     = (const float*)d_in[6];
  const float* eps    = (const float*)d_in[7];
  const float* fW1    = (const float*)d_in[8];
  const float* fb1    = (const float*)d_in[9];
  const float* fW2    = (const float*)d_in[10];
  const float* fb2    = (const float*)d_in[11];
  float* out = (float*)d_out;

  char* base = (char*)d_ws;
  size_t off = 0;
  float*    agg = (float*)(base + off);    off = align256(off + (size_t)NNODES * DCH * 4);
  _Float16* z1  = (_Float16*)(base + off); off = align256(off + (size_t)NNODES * HW * 2);
  float*    h   = (float*)(base + off);    off = align256(off + (size_t)NNODES * DCH * 4);
  _Float16* wp1 = (_Float16*)(base + off); off = align256(off + (size_t)3 * PACK1_PER_LAYER * 2);
  _Float16* wp2 = (_Float16*)(base + off); off = align256(off + (size_t)3 * PACK2_PER_LAYER * 2);
  float*    g   = (float*)(base + off);    off = align256(off + (size_t)NGRAPH * DCH * 4);
  (void)ws_size; (void)n_in; (void)in_sizes; (void)out_size;

  const int packTotal = 3 * (PACK1_PER_LAYER + PACK2_PER_LAYER);
  pack_weights<<<(packTotal + 255) / 256, 256, 0, stream>>>(W1, W2, wp1, wp2);
  zero_g<<<(NGRAPH * DCH + 255) / 256, 256, 0, stream>>>(g);

  const int gemm1Blocks = (MTILES * 2 * 32 + 255) / 256;  // 782 blocks (2 waves/M-tile)
  const int gemm2Blocks = (MTILES * 32 + 255) / 256;      // 391 blocks
  for (int l = 0; l < 3; l++) {
    const float* hin = (l == 0) ? x : h;
    init_agg<<<(NNODES * DCH + 255) / 256, 256, 0, stream>>>(hin, eps, l, agg);
    scatter_edges<<<(NEDGES * 24 + 255) / 256, 256, 0, stream>>>(hin, ei, agg);
    gin_gemm1<<<gemm1Blocks, 256, 0, stream>>>(agg, wp1 + (size_t)l * PACK1_PER_LAYER,
                                               b1 + l * HW, z1);
    gin_gemm2<<<gemm2Blocks, 256, 0, stream>>>(z1, wp2 + (size_t)l * PACK2_PER_LAYER,
                                               b2 + l * DCH, h);
  }

  pool_graphs<<<(NNODES + POOL_NB - 1) / POOL_NB, 256, 0, stream>>>(h, batch, g);
  final_mlp<<<1, 128, 0, stream>>>(g, fW1, fb1, fW2, fb2, out);
}
